// MultiHeadAttention_3994319586027
// MI455X (gfx1250) — matrix operand order repaired and checked by reading, not compiled, not run on hardware
//
#include <hip/hip_runtime.h>
#include <hip/hip_bf16.h>

#define DEV __device__ __forceinline__

typedef __attribute__((ext_vector_type(16))) __bf16        v16bf_t;
typedef __attribute__((ext_vector_type(8)))  float         v8f_t;
typedef __attribute__((ext_vector_type(4)))  unsigned int  uu4;
typedef __attribute__((ext_vector_type(2)))  unsigned int  uu2;
typedef __attribute__((ext_vector_type(4)))  float         ff4;

constexpr int Bc = 4, Sc = 2048, Dc = 1024, Hc = 8, DKc = 128;

DEV unsigned short f2bf_u(float f) {
  unsigned u = __builtin_bit_cast(unsigned, f);
  u += 0x7FFFu + ((u >> 16) & 1u);          // round-to-nearest-even
  return (unsigned short)(u >> 16);
}

union ABu { v16bf_t v; uu4 q[2]; };

// A-matrix 16x32 bf16 (row-major source, row stride ld elements).
// lanes 0-15: M=lane, K = {0..7, 16..23}; lanes 16-31: M=lane-16, K = {8..15, 24..31}
DEV v16bf_t load_a_16x32(const unsigned short* base, int ld, int lane) {
  int m = lane & 15;
  int koff = (lane >> 4) << 3;              // 0 or 8
  const unsigned short* p = base + (size_t)m * ld + koff;
  ABu u;
  u.q[0] = *(const uu4*)(p);                // K = koff .. koff+7   -> VGPR0..3
  u.q[1] = *(const uu4*)(p + 16);           // K = koff+16..koff+23 -> VGPR4..7
  return u.v;
}

// B-matrix 32x16 bf16 from "Bt" storage: Bt[n][k] contiguous in k (row stride ld).
// lanes 0-15: N=lane, K = {0..7, 16..23}; lanes 16-31: N=lane-16, K = {8..15, 24..31}
DEV v16bf_t load_b_32x16(const unsigned short* base, int ld, int lane) {
  int n = lane & 15;
  int kb = (lane >> 4) << 3;                // 0 or 8
  const unsigned short* p = base + (size_t)n * ld + kb;
  ABu u;
  u.q[0] = *(const uu4*)(p);
  u.q[1] = *(const uu4*)(p + 16);
  return u.v;
}

DEV v8f_t wmma_bf16(v16bf_t a, v16bf_t b, v8f_t c) {
  return __builtin_amdgcn_wmma_f32_16x16x32_bf16(false, a, false, b, (short)0, c,
                                                 false, false);
}

DEV float rmax16(float x) {
  x = fmaxf(x, __shfl_xor(x, 1));
  x = fmaxf(x, __shfl_xor(x, 2));
  x = fmaxf(x, __shfl_xor(x, 4));
  x = fmaxf(x, __shfl_xor(x, 8));
  return x;
}
DEV float rsum16(float x) {
  x += __shfl_xor(x, 1);
  x += __shfl_xor(x, 2);
  x += __shfl_xor(x, 4);
  x += __shfl_xor(x, 8);
  return x;
}

// ---------------- prep kernels ----------------
__global__ void k_cvt_x(const float* __restrict__ in, unsigned short* __restrict__ out,
                        int n4) {
  int i = blockIdx.x * blockDim.x + threadIdx.x;
  if (i >= n4) return;
  ff4 f = *(const ff4*)(in + (size_t)i * 4);
  uu2 o;
  o.x = (unsigned)f2bf_u(f.x) | ((unsigned)f2bf_u(f.y) << 16);
  o.y = (unsigned)f2bf_u(f.z) | ((unsigned)f2bf_u(f.w) << 16);
  *(uu2*)(out + (size_t)i * 4) = o;
}

// out[w][h][k][d] = W_w[h][d][k]  (bf16)
__global__ void k_tr_w(const float* __restrict__ WQ, const float* __restrict__ WK,
                       const float* __restrict__ WV, unsigned short* __restrict__ out) {
  long long i = (long long)blockIdx.x * blockDim.x + threadIdx.x;
  int d = (int)(i % Dc);
  long long t = i / Dc;
  int k = (int)(t % DKc); t /= DKc;
  int h = (int)(t % Hc);  t /= Hc;
  int w = (int)t;
  const float* src = (w == 0) ? WQ : (w == 1) ? WK : WV;
  out[i] = f2bf_u(src[((size_t)h * Dc + d) * DKc + k]);
}

// out[k][d] = WO_w[d][k]  (bf16), k in [0,128), d in [0,1024)
__global__ void k_tr_wo(const float* __restrict__ WO, unsigned short* __restrict__ out) {
  int i = blockIdx.x * blockDim.x + threadIdx.x;
  int d = i % Dc;
  int k = i / Dc;
  out[i] = f2bf_u(WO[(size_t)d * DKc + k]);
}

// ---------------- QKV projection ----------------
// wave -> (w,b,h,st): 16 rows x 128 cols of Q/K/V for head h
__global__ void __launch_bounds__(256) k_qkv(
    const unsigned short* __restrict__ xbf, const unsigned short* __restrict__ Wt,
    unsigned short* __restrict__ Qo, unsigned short* __restrict__ Ko,
    unsigned short* __restrict__ Vt) {
  int lane = threadIdx.x & 31;
  int wave = blockIdx.x * (blockDim.x >> 5) + (threadIdx.x >> 5);
  int st = wave & 127;  int t = wave >> 7;
  int h = t & 7;        t >>= 3;
  int b = t & 3;        t >>= 2;
  int w = t;            // 0=Q 1=K 2=V

  const unsigned short* arow  = xbf + ((size_t)b * Sc + st * 16) * Dc;
  const unsigned short* panel = Wt + (size_t)(w * Hc + h) * DKc * Dc;

  v8f_t acc[8] = {};
  for (int kk = 0; kk < Dc; kk += 32) {
    v16bf_t a = load_a_16x32(arow + kk, Dc, lane);
#pragma unroll
    for (int nt = 0; nt < 8; ++nt) {
      v16bf_t bm = load_b_32x16(panel + (size_t)(nt * 16) * Dc + kk, Dc, lane);
      acc[nt] = wmma_bf16(a, bm, acc[nt]);
    }
  }

  int half = lane >> 4, nl = lane & 15;
  if (w < 2) {
    unsigned short* dst =
        (w == 0 ? Qo : Ko) + ((size_t)(b * Hc + h) * Sc + st * 16) * DKc;
#pragma unroll
    for (int nt = 0; nt < 8; ++nt)
#pragma unroll
      for (int r = 0; r < 8; ++r) {
        int m = r + half * 8;
        dst[(size_t)m * DKc + nt * 16 + nl] = f2bf_u(acc[nt][r]);
      }
  } else {
    unsigned short* dst = Vt + (size_t)(b * Hc + h) * DKc * Sc;
#pragma unroll
    for (int nt = 0; nt < 8; ++nt) {
      uu4 pk;  // rows r..r+7 are consecutive s -> contiguous 16B store
      pk.x = (unsigned)f2bf_u(acc[nt][0]) | ((unsigned)f2bf_u(acc[nt][1]) << 16);
      pk.y = (unsigned)f2bf_u(acc[nt][2]) | ((unsigned)f2bf_u(acc[nt][3]) << 16);
      pk.z = (unsigned)f2bf_u(acc[nt][4]) | ((unsigned)f2bf_u(acc[nt][5]) << 16);
      pk.w = (unsigned)f2bf_u(acc[nt][6]) | ((unsigned)f2bf_u(acc[nt][7]) << 16);
      *(uu4*)(dst + (size_t)(nt * 16 + nl) * Sc + st * 16 + half * 8) = pk;
    }
  }
}

// ---------------- flash attention ----------------
__global__ void __launch_bounds__(256) k_attn(
    const unsigned short* __restrict__ Qm, const unsigned short* __restrict__ Km,
    const unsigned short* __restrict__ Vtm, unsigned short* __restrict__ Obf) {
  __shared__ alignas(16) unsigned short Pb[8][16 * 32];  // per-wave P staging
  int lane = threadIdx.x & 31;
  int wib = threadIdx.x >> 5;
  int wave = blockIdx.x * (blockDim.x >> 5) + wib;
  int qt = wave & 127;  int t = wave >> 7;
  int h = t & 7;  int b = (t >> 3) & 3;

  const size_t bh = (size_t)(b * Hc + h);
  const unsigned short* Qbase = Qm + (bh * Sc + (size_t)qt * 16) * DKc;
  const unsigned short* Kbase = Km + bh * Sc * DKc;
  const unsigned short* Vbase = Vtm + bh * DKc * Sc;

  v16bf_t qa[4];
#pragma unroll
  for (int c = 0; c < 4; ++c) qa[c] = load_a_16x32(Qbase + c * 32, DKc, lane);

  v8f_t acc[8] = {};
  float mrow[8], lrow[8];
#pragma unroll
  for (int r = 0; r < 8; ++r) { mrow[r] = -1e30f; lrow[r] = 0.0f; }

  int half = lane >> 4, nl = lane & 15;
  int q0 = qt * 16;
  int nblk = (q0 + 47) >> 5;  // ceil((q0+16)/32)
  unsigned short* Pw = &Pb[wib][0];
  const float scale = 0.08838834764831845f;  // 1/sqrt(128)

  for (int tb = 0; tb < nblk; ++tb) {
    int t0 = tb * 32;
    v8f_t s1 = {}, s2 = {};
#pragma unroll
    for (int c = 0; c < 4; ++c) {
      v16bf_t b1 = load_b_32x16(Kbase + (size_t)t0 * DKc + c * 32, DKc, lane);
      s1 = wmma_bf16(qa[c], b1, s1);
      v16bf_t b2 = load_b_32x16(Kbase + (size_t)(t0 + 16) * DKc + c * 32, DKc, lane);
      s2 = wmma_bf16(qa[c], b2, s2);
    }
    bool needMask = (t0 + 31 > q0);
#pragma unroll
    for (int r = 0; r < 8; ++r) {
      float a1 = s1[r] * scale, a2 = s2[r] * scale;
      if (needMask) {
        int q = q0 + r + half * 8;
        if (t0 + nl > q)      a1 = -1e30f;
        if (t0 + 16 + nl > q) a2 = -1e30f;
      }
      float mx = rmax16(fmaxf(a1, a2));
      float mnew = fmaxf(mrow[r], mx);
      float corr = __expf(mrow[r] - mnew);
      float e1 = __expf(a1 - mnew);
      float e2 = __expf(a2 - mnew);
      lrow[r] = lrow[r] * corr + rsum16(e1 + e2);
      mrow[r] = mnew;
#pragma unroll
      for (int c = 0; c < 8; ++c) acc[c][r] *= corr;
      int m = r + half * 8;
      Pw[m * 32 + nl]      = f2bf_u(e1);
      Pw[m * 32 + 16 + nl] = f2bf_u(e2);
    }
    __asm__ volatile("s_wait_dscnt 0" ::: "memory");
    v16bf_t pa = load_a_16x32(Pw, 32, lane);
#pragma unroll
    for (int c = 0; c < 8; ++c) {
      v16bf_t bv = load_b_32x16(Vbase + (size_t)(c * 16) * Sc + t0, Sc, lane);
      acc[c] = wmma_bf16(pa, bv, acc[c]);
    }
  }

#pragma unroll
  for (int r = 0; r < 8; ++r) {
    float inv = 1.0f / lrow[r];
    int m = r + half * 8;
    size_t row = (size_t)b * Sc + q0 + m;
#pragma unroll
    for (int c = 0; c < 8; ++c)
      Obf[row * (Hc * DKc) + h * DKc + c * 16 + nl] = f2bf_u(acc[c][r] * inv);
  }
}

// ---------------- output projection ----------------
__global__ void __launch_bounds__(256) k_oproj(
    const unsigned short* __restrict__ Obf, const unsigned short* __restrict__ WOt,
    const float* __restrict__ bias, float* __restrict__ out) {
  int lane = threadIdx.x & 31;
  int wave = blockIdx.x * (blockDim.x >> 5) + (threadIdx.x >> 5);  // 0..511
  const unsigned short* arow = Obf + (size_t)wave * 16 * (Hc * DKc);

  v8f_t acc[8] = {};
  for (int kk = 0; kk < Hc * DKc; kk += 32) {
    v16bf_t a = load_a_16x32(arow + kk, Hc * DKc, lane);
#pragma unroll
    for (int nt = 0; nt < 8; ++nt) {
      v16bf_t bm = load_b_32x16(WOt + (size_t)(nt * 16) * (Hc * DKc) + kk,
                                Hc * DKc, lane);
      acc[nt] = wmma_bf16(a, bm, acc[nt]);
    }
  }
  int half = lane >> 4, nl = lane & 15;
#pragma unroll
  for (int nt = 0; nt < 8; ++nt) {
    float bv = bias[nt * 16 + nl];
#pragma unroll
    for (int r = 0; r < 8; ++r) {
      int m = r + half * 8;
      out[((size_t)wave * 16 + m) * DKc + nt * 16 + nl] = acc[nt][r] + bv;
    }
  }
}

extern "C" void kernel_launch(void* const* d_in, const int* in_sizes, int n_in,
                              void* d_out, int out_size, void* d_ws, size_t ws_size,
                              hipStream_t stream) {
  (void)in_sizes; (void)n_in; (void)out_size; (void)ws_size;
  const float* x   = (const float*)d_in[0];
  const float* WQ  = (const float*)d_in[1];
  const float* WK  = (const float*)d_in[2];
  const float* WV  = (const float*)d_in[3];
  const float* WOw = (const float*)d_in[4];
  const float* WOb = (const float*)d_in[5];
  float* out = (float*)d_out;

  unsigned short* xbf = (unsigned short*)d_ws;
  unsigned short* Wt  = xbf + (size_t)Bc * Sc * Dc;            // 3*H*DK*D bf16
  unsigned short* WOt = Wt + (size_t)3 * Hc * DKc * Dc;        // DK x (H*DK)
  unsigned short* Q   = WOt + (size_t)DKc * Hc * DKc;
  unsigned short* K   = Q + (size_t)Bc * Hc * Sc * DKc;
  unsigned short* Vt  = K + (size_t)Bc * Hc * Sc * DKc;
  unsigned short* Obf = Vt + (size_t)Bc * Hc * Sc * DKc;

  k_cvt_x<<<(Bc * Sc * Dc / 4 + 255) / 256, 256, 0, stream>>>(x, xbf, Bc * Sc * Dc / 4);
  k_tr_w<<<(3 * Hc * DKc * Dc) / 256, 256, 0, stream>>>(WQ, WK, WV, Wt);
  k_tr_wo<<<(DKc * Hc * DKc) / 256, 256, 0, stream>>>(WOw, WOt);

  k_qkv<<<(3 * Bc * Hc * (Sc / 16)) / 8, 256, 0, stream>>>(xbf, Wt, Q, K, Vt);
  k_attn<<<(Bc * Hc * (Sc / 16)) / 8, 256, 0, stream>>>(Q, K, Vt, Obf);
  k_oproj<<<(Bc * Sc / 16) / 8, 256, 0, stream>>>(Obf, WOt, WOb, out);
}
